// PatternConv_87617332838729
// MI455X (gfx1250) — matrix operand order repaired and checked by reading, not compiled, not run on hardware
//
#include <hip/hip_runtime.h>
#include <hip/hip_bf16.h>

// PatternConv on MI455X (gfx1250, wave32):
// 64 GEMMs (one per (h,w) pattern pos): D[64 x 2048] = A[64 x 576] * B[576 x 2048] + bias
// bf16 WMMA (v_wmma_f32_16x16x32_bf16), f32 accumulate.
//
// Pass 1 (xpose): x[b][c][y][x] f32 -> xbf[b][y][x][c] bf16 in d_ws.
// Pass 2 (main): WG = 256 threads (8 waves) = one (h,w) and a 512-pixel N tile.
//   A staged once in LDS in WMMA fragment order (72KB);
//   B double-buffered (2x32KB), staged for ks+1 with GLOBAL_LOAD_ASYNC_TO_LDS_B128
//   (ASYNCcnt) overlapping the 16 WMMAs of step ks; s_wait_asynccnt before barrier.

typedef __attribute__((ext_vector_type(16))) __bf16 v16bf;
typedef __attribute__((ext_vector_type(8)))  __bf16 v8bf;
typedef __attribute__((ext_vector_type(8)))  float  v8f;

#define KW 3
#define Hn 128
#define Wn 128
#define On 64
#define KTOT 576            // Cn * KW * KW
#define KSTEPS 18           // KTOT / 32
#define NTILE 512           // pixels per workgroup
#define NT_PER_WG 32        // NTILE / 16
#define XBF_BYTES (8ull * 128 * 128 * 64 * 2)   // 16.78 MB

#define ASYNC_LDS_B128(ldsoff, gaddr)                                          \
    asm volatile("global_load_async_to_lds_b128 %0, %1, off"                   \
                 :: "v"(ldsoff), "v"(gaddr) : "memory")
#define ASYNC_LDS_B128_O16(ldsoff, gaddr)                                      \
    asm volatile("global_load_async_to_lds_b128 %0, %1, off offset:16"         \
                 :: "v"(ldsoff), "v"(gaddr) : "memory")
#define WAIT_ASYNCCNT0() asm volatile("s_wait_asynccnt 0x0" ::: "memory")

// ---------------- Pass 1: transpose + convert x -> xbf[b][y][x][c] ----------------
__global__ __launch_bounds__(256) void xpose_kernel(const float* __restrict__ x,
                                                    __bf16* __restrict__ xbf)
{
    // grid: (b, y, x-tile of 32): 8*128*4 blocks
    const int blk = blockIdx.x;
    const int xt  = blk & 3;
    const int by  = blk >> 2;
    const int y   = by & 127;
    const int b   = by >> 7;
    const int x0  = xt * 32;

    __shared__ __bf16 t[32][72];   // [x_local][c], padded row

    const int tid = threadIdx.x;
    const int xl = tid & 31;
    const int cg = tid >> 5;       // 8 channel groups of 8
    const float* src = x + ((size_t)b << 20) + ((size_t)y << 7) + x0 + xl;
#pragma unroll
    for (int cc = 0; cc < 8; ++cc) {
        int c = cg * 8 + cc;
        t[xl][c] = (__bf16)src[(size_t)c << 14];
    }
    __syncthreads();
    const int cg2 = tid & 7;
    const int xl2 = tid >> 3;
    v8bf vals;
#pragma unroll
    for (int cc = 0; cc < 8; ++cc) vals[cc] = t[xl2][cg2 * 8 + cc];
    __bf16* dst = xbf + ((((size_t)b * 128 + y) * 128) + x0 + xl2) * 64 + cg2 * 8;
    *(v8bf*)dst = vals;
}

// ---------------- B staging: fast path, async Global->LDS ----------------
// B-fragment order: flat idx = (nt*32 + lane)*16 + e ; lane -> N = lane&15,
// k = e + 16*(lane>=16). For pixel n the 32 k-values are 64 contiguous bytes.
__device__ __forceinline__ void stage_B_fast(const __bf16* __restrict__ xbf,
                                             __bf16* __restrict__ buf,
                                             int ks, int h, int w, int nb4, int tid)
{
    const int uv    = ks >> 1;
    const int uu    = uv / 3;
    const int vv    = uv % 3;
    const int chalf = ks & 1;
#pragma unroll
    for (int half = 0; half < 2; ++half) {
        const int n    = tid + half * 256;       // pixel within tile: 0..511
        const int nt   = n >> 4;
        const int jn   = n & 15;
        const int idxp = nb4 + nt;               // 0..127
        const int b    = idxp >> 4;
        const int i    = idxp & 15;
        const int y    = 8 * i + h + uu;
        const int xx   = 8 * jn + w + vv;
        __bf16* lo = buf + ((nt * 32) + jn) * 16;        // k 0..15  (32B)
        __bf16* hi = buf + ((nt * 32) + jn + 16) * 16;   // k 16..31 (32B)
        if (y < Hn && xx < Wn) {
            const __bf16* src = xbf +
                (((((size_t)b << 7) + y) << 7) + xx) * 64 + chalf * 32;
            unsigned lds_lo = (unsigned)(uintptr_t)lo;
            unsigned lds_hi = (unsigned)(uintptr_t)hi;
            unsigned long long g_lo = (unsigned long long)(uintptr_t)src;
            unsigned long long g_hi = g_lo + 32;
            ASYNC_LDS_B128(lds_lo, g_lo);          // k 0..7
            ASYNC_LDS_B128_O16(lds_lo, g_lo);      // k 8..15  (+16B lds & global)
            ASYNC_LDS_B128(lds_hi, g_hi);          // k 16..23
            ASYNC_LDS_B128_O16(lds_hi, g_hi);      // k 24..31
        } else {
            v8bf z = (v8bf)(__bf16)0.0f;
            *(v8bf*)lo       = z;
            *(v8bf*)(lo + 8) = z;
            *(v8bf*)hi       = z;
            *(v8bf*)(hi + 8) = z;
        }
    }
}

// ---------------- B staging: fallback direct gather from f32 x ----------------
__device__ __forceinline__ void stage_B_slow(const float* __restrict__ x,
                                             __bf16* __restrict__ buf,
                                             int ks, int h, int w, int nb4, int tid)
{
    const int eo    = tid & 1;
    const int j     = (tid >> 1) & 15;
    const int g     = (tid >> 5) & 1;
    const int ntq   = tid >> 6;
    const int uv    = ks >> 1;
    const int uu    = uv / 3;
    const int vv    = uv % 3;
    const int cbase = (ks & 1) * 32;
    const int  c0  = cbase + 16 * g + 8 * eo;
    const int  xx  = 8 * j + w + vv;
    const bool xok = (xx < Wn);
    const float* px = x + ((size_t)c0 << 14) + xx;
#pragma unroll
    for (int t = 0; t < 8; ++t) {
        const int nt   = ntq * 8 + t;
        const int idxp = nb4 + nt;
        const int b    = idxp >> 4;
        const int i    = idxp & 15;
        const int y    = 8 * i + h + uu;
        const bool ok  = xok && (y < Hn);
        const float* pp = px + ((size_t)b << 20) + ((size_t)y << 7);
        v8bf vals;
#pragma unroll
        for (int cc = 0; cc < 8; ++cc) {
            float f = ok ? pp[(size_t)cc << 14] : 0.0f;
            vals[cc] = (__bf16)f;
        }
        *(v8bf*)(buf + (((nt * 32) + j + 16 * g) * 16 + 8 * eo)) = vals;
    }
}

// ---------------- Pass 2: WMMA GEMM ----------------
__global__ __launch_bounds__(256) void pattern_conv_wmma(
    const float* __restrict__ x,      // [B, C, H, W] f32
    const __bf16* __restrict__ xbf,   // [B, H, W, C] bf16 (valid iff fast)
    const float* __restrict__ wgt,    // [HS, WS, O, C, 3, 3]
    const float* __restrict__ bias,   // [HS, WS, O]
    float* __restrict__ out,          // [B, O, H, W]
    int fast)
{
    __shared__ v16bf Alds[4 * KSTEPS * 32];        // 73728 B
    __shared__ v16bf Blds[2][NT_PER_WG * 32];      // 65536 B

    const int tid  = threadIdx.x;
    const int lane = tid & 31;
    const int wave = tid >> 5;
    const int mt   = wave & 3;   // M tile (16 rows of O)
    const int nh   = wave >> 2;  // N half (16 n-tiles each)

    const int hw     = blockIdx.x >> 2;       // pattern position 0..63
    const int h      = hw >> 3;
    const int w      = hw & 7;
    const int nq     = blockIdx.x & 3;
    const int n_base = nq * NTILE;
    const int nb4    = nq * 32;

    const float* wsrc = wgt + hw * (On * KTOT);

    // ---- Stage A (weights) into LDS in WMMA A-fragment order (bf16), once ----
    // A 16x32 as staged here (contiguous k, same as B): lane L holds row M=L&15;
//   elem e -> K = e + 16*(L>=16)
    for (int idx = tid; idx < 4 * KSTEPS * 32 * 16; idx += 256) {
        int e    = idx & 15;
        int ln   = (idx >> 4) & 31;
        int rem  = idx >> 9;
        int ks   = rem % KSTEPS;
        int mti  = rem / KSTEPS;
        int r    = e >> 1;
        int kb   = (r < 4) ? (2 * r) : (16 + 2 * (r - 4));
        int k    = e + 16 * (ln >> 4);
        int uv   = ks >> 1;
        int c    = (ks & 1) * 32 + k;
        int o    = mti * 16 + (ln & 15);
        ((__bf16*)Alds)[idx] = (__bf16)wsrc[o * KTOT + c * (KW * KW) + uv];
    }

    if (fast) stage_B_fast(xbf, (__bf16*)Blds[0], 0, h, w, nb4, tid);
    else      stage_B_slow(x,   (__bf16*)Blds[0], 0, h, w, nb4, tid);
    WAIT_ASYNCCNT0();
    __syncthreads();

    v8f acc[16];
#pragma unroll
    for (int t = 0; t < 16; ++t) acc[t] = (v8f)0.0f;

    for (int ks = 0; ks < KSTEPS; ++ks) {
        if (ks + 1 < KSTEPS) {
            if (fast) stage_B_fast(xbf, (__bf16*)Blds[(ks + 1) & 1], ks + 1, h, w, nb4, tid);
            else      stage_B_slow(x,   (__bf16*)Blds[(ks + 1) & 1], ks + 1, h, w, nb4, tid);
        }
        v16bf afrag = Alds[(mt * KSTEPS + ks) * 32 + lane];
        const v16bf* bbuf = Blds[ks & 1];
#pragma unroll
        for (int t = 0; t < 16; ++t) {
            v16bf bfrag = bbuf[(nh * 16 + t) * 32 + lane];
            acc[t] = __builtin_amdgcn_wmma_f32_16x16x32_bf16(
                false, afrag, false, bfrag, (short)0, acc[t], false, false);
        }
        WAIT_ASYNCCNT0();       // staged tile for ks+1 fully in LDS
        __syncthreads();
    }

    // ---- Epilogue: bias + scatter to [b, o, 8i+h, 8j+w] ----
    // C/D layout: VGPR r, lane L -> M = r + 8*(L>=16), N = L&15
    const int obase = mt * 16 + 8 * (lane >> 4);
    float bv[8];
#pragma unroll
    for (int r = 0; r < 8; ++r) bv[r] = bias[hw * On + obase + r];

#pragma unroll
    for (int t = 0; t < 16; ++t) {
        int p = n_base + (nh * 16 + t) * 16 + (lane & 15);
        int b = p >> 8;
        int i = (p >> 4) & 15;
        int j = p & 15;
        int yy = 8 * i + h;
        int xx = 8 * j + w;
        float* op = out + ((size_t)b << 20) + (yy << 7) + xx;
#pragma unroll
        for (int r = 0; r < 8; ++r)
            op[(size_t)(obase + r) << 14] = acc[t][r] + bv[r];
    }
}

extern "C" void kernel_launch(void* const* d_in, const int* in_sizes, int n_in,
                              void* d_out, int out_size, void* d_ws, size_t ws_size,
                              hipStream_t stream) {
    const float* x    = (const float*)d_in[0];
    const float* wgt  = (const float*)d_in[1];
    const float* bias = (const float*)d_in[2];
    float* out        = (float*)d_out;
    (void)in_sizes; (void)n_in; (void)out_size;

    const int fast = (d_ws != nullptr && ws_size >= XBF_BYTES) ? 1 : 0;
    __bf16* xbf = (__bf16*)d_ws;

    if (fast) {
        dim3 g1(8 * 128 * 4), b1(256);
        hipLaunchKernelGGL(xpose_kernel, g1, b1, 0, stream, x, xbf);
    }
    dim3 g2(64 * 4), b2(256);
    hipLaunchKernelGGL(pattern_conv_wmma, g2, b2, 0, stream,
                       x, xbf, wgt, bias, out, fast);
}
